// SparselyRoutedMLP_75737453297875
// MI455X (gfx1250) — matrix operand order repaired and checked by reading, not compiled, not run on hardware
//
#include <hip/hip_runtime.h>
#include <stdint.h>

// Problem dims (fixed by reference): B=2, S=2048, D=1024, H=2048, E=8, k=2
#define D_DIM 1024
#define H_DIM 2048
#define E_NUM 8
#define NTOK  4096   // B*S
#define TM    32     // tokens per expert-kernel block
#define MAXT  NTOK   // per-expert list capacity (worst case)

#define XS_STR (D_DIM + 8)   // padded LDS row stride (elements) -> bank-friendly b128
#define HS_STR (H_DIM + 8)

typedef __attribute__((ext_vector_type(16))) __bf16   v16bf;
typedef __attribute__((ext_vector_type(8)))  float    v8f;
typedef __attribute__((ext_vector_type(8)))  uint32_t u32x8;

__device__ __forceinline__ uint32_t f2bf(float f) {
    uint32_t u = __builtin_bit_cast(uint32_t, f);
    uint32_t r = (u + 0x7FFFu + ((u >> 16) & 1u)) >> 16;   // RNE
    return r & 0xFFFFu;
}
__device__ __forceinline__ uint32_t pack2bf(float lo, float hi) {
    return f2bf(lo) | (f2bf(hi) << 16);
}
__device__ __forceinline__ v16bf mkfrag(uint4 lo, uint4 hi) {
    u32x8 u;
    u[0] = lo.x; u[1] = lo.y; u[2] = lo.z; u[3] = lo.w;
    u[4] = hi.x; u[5] = hi.y; u[6] = hi.z; u[7] = hi.w;
    return __builtin_bit_cast(v16bf, u);
}

// ---------------------------------------------------------------------------
// Weight pre-swizzle: fp32 [E][K][N] -> bf16 in per-lane WMMA B-fragment order.
// For K-tile kt, N-tile nt, lane (half,n), reg v: elements B[kt*32+half*16+2v+{0,1}][nt*16+n]
// packed as one uint at dst[((tileAll*32) + half*16 + n)*8 + v].
// ---------------------------------------------------------------------------
template <int K, int N>
__global__ void __launch_bounds__(256)
swizzle_weights_kernel(const float* __restrict__ src, uint32_t* __restrict__ dst) {
    constexpr int NT      = N / 16;
    constexpr int TILES_E = (K / 32) * NT;
    size_t g = (size_t)blockIdx.x * 256 + threadIdx.x;  // < E*K*N/2

    int n    = (int)(g & 15);
    int v    = (int)((g >> 4) & 7);
    int half = (int)((g >> 7) & 1);
    size_t tileAll = g >> 8;

    int e  = (int)(tileAll / TILES_E);
    int t  = (int)(tileAll % TILES_E);
    int kt = t / NT;
    int nt = t % NT;

    int k0  = kt * 32 + half * 16 + 2 * v;
    int col = nt * 16 + n;

    const float* srcE = src + (size_t)e * K * N;
    float w0 = srcE[(size_t)k0 * N + col];
    float w1 = srcE[(size_t)(k0 + 1) * N + col];

    size_t outIdx = (tileAll * 32 + (size_t)(half * 16 + n)) * 8 + v;
    dst[outIdx] = pack2bf(w0, w1);
}

// ---------------------------------------------------------------------------
// Router: logits = x @ Wr + br ; top-2 ; softmax over the 2 ; append to lists
// ---------------------------------------------------------------------------
__global__ void __launch_bounds__(256)
moe_router_kernel(const float* __restrict__ x,
                  const float* __restrict__ Wr,
                  const float* __restrict__ br,
                  int*   __restrict__ counts,
                  int*   __restrict__ tokList,
                  float* __restrict__ scrList) {
    __shared__ float red[256][E_NUM];
    const int t   = blockIdx.x;
    const int tid = threadIdx.x;
    const float* xr = x + (size_t)t * D_DIM;

    float acc[E_NUM];
#pragma unroll
    for (int e = 0; e < E_NUM; ++e) acc[e] = 0.0f;

    for (int d = tid; d < D_DIM; d += 256) {
        float xv = xr[d];
#pragma unroll
        for (int e = 0; e < E_NUM; ++e) acc[e] += xv * Wr[d * E_NUM + e];
    }
#pragma unroll
    for (int e = 0; e < E_NUM; ++e) red[tid][e] = acc[e];
    __syncthreads();

    for (int s = 128; s > 0; s >>= 1) {
        if (tid < s) {
#pragma unroll
            for (int e = 0; e < E_NUM; ++e) red[tid][e] += red[tid + s][e];
        }
        __syncthreads();
    }

    if (tid == 0) {
        float lg[E_NUM];
#pragma unroll
        for (int e = 0; e < E_NUM; ++e) lg[e] = red[0][e] + br[e];
        int i0 = 0;
#pragma unroll
        for (int e = 1; e < E_NUM; ++e) if (lg[e] > lg[i0]) i0 = e;
        int i1 = (i0 == 0) ? 1 : 0;
#pragma unroll
        for (int e = 0; e < E_NUM; ++e)
            if (e != i0 && lg[e] > lg[i1]) i1 = e;

        float e1  = __expf(lg[i1] - lg[i0]);
        float inv = 1.0f / (1.0f + e1);

        int pos0 = atomicAdd(&counts[i0], 1);
        tokList[i0 * MAXT + pos0] = t;
        scrList[i0 * MAXT + pos0] = inv;
        int pos1 = atomicAdd(&counts[i1], 1);
        tokList[i1 * MAXT + pos1] = t;
        scrList[i1 * MAXT + pos1] = e1 * inv;
    }
}

// ---------------------------------------------------------------------------
// Expert MLP: per block, 32 tokens of one expert.
//   GEMM1: H = relu(X[32x1024] @ W1[e] + b1)   (bf16 WMMA, H tile in LDS)
//   GEMM2: out += score * (H @ W2[e] + b2)     (atomic scatter-add combine)
// K-loop software-pipelined (B frags prefetched 1 step ahead); each wave
// processes 2 N-tiles x 2 M-tiles -> 4 independent WMMA chains sharing A frags.
// ---------------------------------------------------------------------------
__global__ void __launch_bounds__(256)
moe_expert_kernel(const float*    __restrict__ x,
                  const uint32_t* __restrict__ W1s,
                  const float*    __restrict__ b1,
                  const uint32_t* __restrict__ W2s,
                  const float*    __restrict__ b2,
                  const int*      __restrict__ counts,
                  const int*      __restrict__ tokList,
                  const float*    __restrict__ scrList,
                  float*          __restrict__ out) {
    extern __shared__ char smem[];
    uint16_t* Xs   = (uint16_t*)smem;                               // [TM][XS_STR] bf16
    uint16_t* Hs   = (uint16_t*)(smem + (size_t)TM * XS_STR * 2);   // [TM][HS_STR] bf16
    int*      tokS = (int*)(smem + (size_t)TM * XS_STR * 2 + (size_t)TM * HS_STR * 2);
    float*    scrS = (float*)(tokS + TM);

    const int e    = blockIdx.y;
    const int cnt  = counts[e];
    const int base = blockIdx.x * TM;
    if (base >= cnt) return;
    int rows = cnt - base;
    if (rows > TM) rows = TM;

    const int tid = threadIdx.x;

    if (tid < TM) {
        int ok = (tid < rows);
        tokS[tid] = ok ? tokList[e * MAXT + base + tid] : 0;
        scrS[tid] = ok ? scrList[e * MAXT + base + tid] : 0.0f;
    }
    __syncthreads();

    // ---- gather X tile, fp32 -> bf16, into LDS (zero padding rows) ----
    for (int idx = tid; idx < TM * (D_DIM / 4); idx += 256) {
        int row = idx / (D_DIM / 4);
        int c4  = idx % (D_DIM / 4);
        float4 v = make_float4(0.f, 0.f, 0.f, 0.f);
        if (row < rows) {
            const float4* src = (const float4*)(x + (size_t)tokS[row] * D_DIM);
            v = src[c4];
        }
        uint32_t* dst = (uint32_t*)(Xs + (size_t)row * XS_STR + c4 * 4);
        dst[0] = pack2bf(v.x, v.y);
        dst[1] = pack2bf(v.z, v.w);
    }
    __syncthreads();

    const int wave = tid >> 5;
    const int lane = tid & 31;
    const int half = lane >> 4;
    const int lm   = lane & 15;

    // =================== GEMM1: H = relu(X @ W1[e] + b1) ===================
    {
        const uint32_t* W1e = W1s + (size_t)e * (D_DIM * H_DIM / 2);
        const float*    b1e = b1 + (size_t)e * H_DIM;
        constexpr int NT = H_DIM / 16;                       // 128
        constexpr int KT = D_DIM / 32;                       // 32
        const uint16_t* a0p = Xs + (size_t)lm * XS_STR + half * 16;
        const uint16_t* a1p = Xs + (size_t)(16 + lm) * XS_STR + half * 16;

        for (int nt = wave * 2; nt < NT; nt += 16) {
            const int ncol0 = nt * 16 + lm;
            const int ncol1 = ncol0 + 16;
            v8f c00 = {}, c01 = {}, c10 = {}, c11 = {};

            // prologue: B frags for kt=0 (two N-tiles)
            const uint4* bp0 = (const uint4*)(W1e + (((size_t)nt) * 32 + lane) * 8);
            const uint4* bp1 = (const uint4*)(W1e + (((size_t)nt + 1) * 32 + lane) * 8);
            uint4 b0l = bp0[0], b0h = bp0[1];
            uint4 b1l = bp1[0], b1h = bp1[1];

            for (int kt = 0; kt < KT; ++kt) {
                const int kb = kt * 32;
                const int kn = (kt + 1 < KT) ? kt + 1 : kt;   // prefetch next (or redo last)
                const uint4* np0 =
                    (const uint4*)(W1e + (((size_t)kn * NT + nt) * 32 + lane) * 8);
                const uint4* np1 =
                    (const uint4*)(W1e + (((size_t)kn * NT + nt + 1) * 32 + lane) * 8);
                uint4 nb0l = np0[0], nb0h = np0[1];
                uint4 nb1l = np1[0], nb1h = np1[1];

                uint4 xa0 = *(const uint4*)(a0p + kb);
                uint4 xa1 = *(const uint4*)(a0p + kb + 8);
                uint4 ya0 = *(const uint4*)(a1p + kb);
                uint4 ya1 = *(const uint4*)(a1p + kb + 8);
                v16bf af0 = mkfrag(xa0, xa1);
                v16bf af1 = mkfrag(ya0, ya1);
                v16bf bf0 = mkfrag(b0l, b0h);
                v16bf bf1 = mkfrag(b1l, b1h);

                c00 = __builtin_amdgcn_wmma_f32_16x16x32_bf16(
                          false, af0, false, bf0, (short)0, c00, false, false);
                c10 = __builtin_amdgcn_wmma_f32_16x16x32_bf16(
                          false, af1, false, bf0, (short)0, c10, false, false);
                c01 = __builtin_amdgcn_wmma_f32_16x16x32_bf16(
                          false, af0, false, bf1, (short)0, c01, false, false);
                c11 = __builtin_amdgcn_wmma_f32_16x16x32_bf16(
                          false, af1, false, bf1, (short)0, c11, false, false);

                b0l = nb0l; b0h = nb0h; b1l = nb1l; b1h = nb1h;
            }
            float bias0 = b1e[ncol0];
            float bias1 = b1e[ncol1];
#pragma unroll
            for (int r = 0; r < 8; ++r) {        // C/D layout: M = r + half*8, N = lm
                int row = r + half * 8;
                float h;
                h = c00[r] + bias0; h = h > 0.f ? h : 0.f;
                Hs[(size_t)row * HS_STR + ncol0] = (uint16_t)f2bf(h);
                h = c10[r] + bias0; h = h > 0.f ? h : 0.f;
                Hs[(size_t)(16 + row) * HS_STR + ncol0] = (uint16_t)f2bf(h);
                h = c01[r] + bias1; h = h > 0.f ? h : 0.f;
                Hs[(size_t)row * HS_STR + ncol1] = (uint16_t)f2bf(h);
                h = c11[r] + bias1; h = h > 0.f ? h : 0.f;
                Hs[(size_t)(16 + row) * HS_STR + ncol1] = (uint16_t)f2bf(h);
            }
        }
    }
    __syncthreads();

    // ============ GEMM2: out += score * (H @ W2[e] + b2) ============
    {
        const uint32_t* W2e = W2s + (size_t)e * (H_DIM * D_DIM / 2);
        const float*    b2e = b2 + (size_t)e * D_DIM;
        constexpr int NT = D_DIM / 16;                       // 64
        constexpr int KT = H_DIM / 32;                       // 64
        const uint16_t* a0p = Hs + (size_t)lm * HS_STR + half * 16;
        const uint16_t* a1p = Hs + (size_t)(16 + lm) * HS_STR + half * 16;

        for (int nt = wave * 2; nt < NT; nt += 16) {
            const int ncol0 = nt * 16 + lm;
            const int ncol1 = ncol0 + 16;
            v8f c00 = {}, c01 = {}, c10 = {}, c11 = {};

            const uint4* bp0 = (const uint4*)(W2e + (((size_t)nt) * 32 + lane) * 8);
            const uint4* bp1 = (const uint4*)(W2e + (((size_t)nt + 1) * 32 + lane) * 8);
            uint4 b0l = bp0[0], b0h = bp0[1];
            uint4 b1l = bp1[0], b1h = bp1[1];

            for (int kt = 0; kt < KT; ++kt) {
                const int kb = kt * 32;
                const int kn = (kt + 1 < KT) ? kt + 1 : kt;
                const uint4* np0 =
                    (const uint4*)(W2e + (((size_t)kn * NT + nt) * 32 + lane) * 8);
                const uint4* np1 =
                    (const uint4*)(W2e + (((size_t)kn * NT + nt + 1) * 32 + lane) * 8);
                uint4 nb0l = np0[0], nb0h = np0[1];
                uint4 nb1l = np1[0], nb1h = np1[1];

                uint4 xa0 = *(const uint4*)(a0p + kb);
                uint4 xa1 = *(const uint4*)(a0p + kb + 8);
                uint4 ya0 = *(const uint4*)(a1p + kb);
                uint4 ya1 = *(const uint4*)(a1p + kb + 8);
                v16bf af0 = mkfrag(xa0, xa1);
                v16bf af1 = mkfrag(ya0, ya1);
                v16bf bf0 = mkfrag(b0l, b0h);
                v16bf bf1 = mkfrag(b1l, b1h);

                c00 = __builtin_amdgcn_wmma_f32_16x16x32_bf16(
                          false, af0, false, bf0, (short)0, c00, false, false);
                c10 = __builtin_amdgcn_wmma_f32_16x16x32_bf16(
                          false, af1, false, bf0, (short)0, c10, false, false);
                c01 = __builtin_amdgcn_wmma_f32_16x16x32_bf16(
                          false, af0, false, bf1, (short)0, c01, false, false);
                c11 = __builtin_amdgcn_wmma_f32_16x16x32_bf16(
                          false, af1, false, bf1, (short)0, c11, false, false);

                b0l = nb0l; b0h = nb0h; b1l = nb1l; b1h = nb1h;
            }
            float bias0 = b2e[ncol0];
            float bias1 = b2e[ncol1];
#pragma unroll
            for (int r = 0; r < 8; ++r) {
                int row0 = r + half * 8;
                int row1 = 16 + row0;
                if (row0 < rows) {
                    float s = scrS[row0];
                    atomicAdd(&out[(size_t)tokS[row0] * D_DIM + ncol0], (c00[r] + bias0) * s);
                    atomicAdd(&out[(size_t)tokS[row0] * D_DIM + ncol1], (c01[r] + bias1) * s);
                }
                if (row1 < rows) {
                    float s = scrS[row1];
                    atomicAdd(&out[(size_t)tokS[row1] * D_DIM + ncol0], (c10[r] + bias0) * s);
                    atomicAdd(&out[(size_t)tokS[row1] * D_DIM + ncol1], (c11[r] + bias1) * s);
                }
            }
        }
    }
}

// ---------------------------------------------------------------------------
extern "C" void kernel_launch(void* const* d_in, const int* in_sizes, int n_in,
                              void* d_out, int out_size, void* d_ws, size_t ws_size,
                              hipStream_t stream) {
    const float* x  = (const float*)d_in[0];
    const float* Wr = (const float*)d_in[1];
    const float* br = (const float*)d_in[2];
    const float* W1 = (const float*)d_in[3];
    const float* b1 = (const float*)d_in[4];
    const float* W2 = (const float*)d_in[5];
    const float* b2 = (const float*)d_in[6];
    float* out = (float*)d_out;

    // workspace layout
    char* ws = (char*)d_ws;
    int*      counts  = (int*)ws;                                       // 256 B
    int*      tokList = (int*)(ws + 256);                               // 128 KB
    float*    scrList = (float*)(ws + 256 + (size_t)E_NUM * MAXT * 4);  // 128 KB
    size_t    wOff    = 256 + (size_t)E_NUM * MAXT * 8;                 // 262400 (256-aligned)
    uint32_t* W1sw    = (uint32_t*)(ws + wOff);                         // 32 MB
    uint32_t* W2sw    = (uint32_t*)(ws + wOff + (size_t)E_NUM * D_DIM * H_DIM * 2);

    hipMemsetAsync(counts, 0, 256, stream);
    hipMemsetAsync(out, 0, (size_t)out_size * sizeof(float), stream);

    // pre-swizzle weights to bf16 WMMA-fragment layout (one pass, L2-resident after)
    {
        size_t n1 = (size_t)E_NUM * D_DIM * H_DIM / 2;   // packed uints
        swizzle_weights_kernel<D_DIM, H_DIM>
            <<<(unsigned)(n1 / 256), 256, 0, stream>>>(W1, W1sw);
        size_t n2 = (size_t)E_NUM * H_DIM * D_DIM / 2;
        swizzle_weights_kernel<H_DIM, D_DIM>
            <<<(unsigned)(n2 / 256), 256, 0, stream>>>(W2, W2sw);
    }

    moe_router_kernel<<<NTOK, 256, 0, stream>>>(x, Wr, br, counts, tokList, scrList);

    size_t smem = (size_t)TM * XS_STR * 2    // Xs bf16 (padded rows)
                + (size_t)TM * HS_STR * 2    // Hs bf16 (padded rows)
                + (size_t)TM * 4             // tokS
                + (size_t)TM * 4;            // scrS   -> ~193.4 KB of the 320 KB WGP LDS
    dim3 grid(NTOK / TM, E_NUM);
    moe_expert_kernel<<<grid, 256, smem, stream>>>(x, W1sw, b1, W2sw, b2,
                                                   counts, tokList, scrList, out);
}
